// MultiHeadAttention_16827681865852
// MI455X (gfx1250) — compile-verified, matrix operand order checked by reading, not run on hardware
//
#include <hip/hip_runtime.h>
#include <hip/hip_bf16.h>
#include <math.h>

// Problem constants (from reference)
#define B_  4
#define Q_  2048
#define K_  2048
#define D_  1024
#define H_  16
#define DK_ 64

typedef __attribute__((ext_vector_type(16))) __bf16 v16bf;
typedef __attribute__((ext_vector_type(8)))  __bf16 v8bf;
typedef __attribute__((ext_vector_type(8)))  float  v8f;

// Native f32->bf16 (RNE); gfx1250 has HW bf16 converts, avoids ALU emulation.
__device__ inline __bf16 f2bf(float f) { return (__bf16)f; }

// ---------------------------------------------------------------------------
// Async global->LDS copies (CDNA5 GLOBAL_LOAD_ASYNC_TO_LDS_B128, ASYNCcnt).
// Probe result (round 2 stderr): builtin takes 4 args, arg0 is
// 'int __attribute__((vector_size(16))) __device__ *' (int4 in AS1).
// ---------------------------------------------------------------------------
#if defined(__gfx1250__) && __has_builtin(__builtin_amdgcn_global_load_async_to_lds_b128)
#define HAVE_ASYNC_LDS 1
#else
#define HAVE_ASYNC_LDS 0
#endif

#if HAVE_ASYNC_LDS
typedef int v4i_vs __attribute__((vector_size(16)));
typedef __attribute__((address_space(1))) v4i_vs* g128_t;
typedef __attribute__((address_space(3))) v4i_vs* l128_t;
__device__ inline void async_cp16(void* lds_dst, const void* glb_src) {
  __builtin_amdgcn_global_load_async_to_lds_b128(
      (g128_t)(v4i_vs*)(void*)glb_src,
      (l128_t)(v4i_vs*)lds_dst,
      0, 0);
}
#endif

__device__ inline void wait_async0() {
#if __has_builtin(__builtin_amdgcn_s_wait_asynccnt)
  __builtin_amdgcn_s_wait_asynccnt(0);
#else
  asm volatile("s_wait_asynccnt 0x0" ::: "memory");
#endif
}

// Load one 16x32 bf16 WMMA fragment (A-layout; B uses the same layout when the
// tile is stored contraction-major). Per ISA 7.12.2: lanes 0-15 hold K=0..7 and
// K=16..23, lanes 16-31 hold K=8..15 and K=24..31, row/col = lane & 15.
__device__ inline v16bf load_frag(const __bf16* base, int ld) {
  int lane = threadIdx.x & 31;
  const __bf16* p = base + (lane & 15) * ld + ((lane >> 4) << 3);
  v8bf lo = *(const v8bf*)(p);        // K = lanehalf*8 + 0..7
  v8bf hi = *(const v8bf*)(p + 16);   // K = 16 + lanehalf*8 + 0..7
  return __builtin_shufflevector(lo, hi, 0,1,2,3,4,5,6,7,8,9,10,11,12,13,14,15);
}

__device__ inline v8f wmma_bf16(v16bf a, v16bf b, v8f c) {
  return __builtin_amdgcn_wmma_f32_16x16x32_bf16(false, a, false, b, (short)0, c,
                                                 false, false);
}

// ---------------------------------------------------------------------------
// Kernel 1: fused Q/K/V projection.  out[b,h,s,e] = sum_d X[b,s,d] * W[h,d,e]
// X: [B*S, D] fp32.  W: [H, D, DK] fp32.  Out: [B, H, S, DK] bf16.
// blockIdx.z selects (query,Wq) / (key,Wk) / (value,Wv); q gets 1/sqrt(DK).
// ---------------------------------------------------------------------------
__global__ __launch_bounds__(256) void qkv_proj_kernel(
    const float* __restrict__ query, const float* __restrict__ keym,
    const float* __restrict__ valuem,
    const float* __restrict__ Wq, const float* __restrict__ Wk,
    const float* __restrict__ Wv,
    __bf16* __restrict__ q_ws, __bf16* __restrict__ k_ws,
    __bf16* __restrict__ v_ws) {
  __shared__ __align__(16) __bf16 Asl[128][40];   // [row][k]
  __shared__ __align__(16) __bf16 Bsl[128][40];   // [col][k]  (contraction-major)

  const int z = blockIdx.z;
  const float* X  = (z == 0) ? query : (z == 1) ? keym : valuem;
  const float* W  = (z == 0) ? Wq    : (z == 1) ? Wk   : Wv;
  __bf16*      Ot = (z == 0) ? q_ws  : (z == 1) ? k_ws : v_ws;
  const float scale = (z == 0) ? 0.125f : 1.0f;   // 1/sqrt(64) folded into q

  const int m0 = blockIdx.x * 128;
  const int n0 = blockIdx.y * 128;
  const int tid = threadIdx.x;
  const int lane = tid & 31;
  const int wid = tid >> 5;
  const int wm = wid & 3;          // 4 waves down M (32 rows each)
  const int wn = wid >> 2;         // 2 waves across N (64 cols each)

  v8f acc[2][4];
  for (int i = 0; i < 2; ++i)
    for (int j = 0; j < 4; ++j) acc[i][j] = (v8f)0.0f;

  for (int kt = 0; kt < D_ / 32; ++kt) {
    __syncthreads();
    {   // stage A tile 128x32 (fp32 -> bf16), vectorized: 2x b128 LDS stores
      int row  = tid >> 1;
      int half = (tid & 1) * 16;
      const float* src = X + (size_t)(m0 + row) * D_ + kt * 32 + half;
#pragma unroll
      for (int g = 0; g < 2; ++g) {
        float4 f0 = *(const float4*)(src + g * 8);
        float4 f1 = *(const float4*)(src + g * 8 + 4);
        v8bf t;
        t[0] = f2bf(f0.x); t[1] = f2bf(f0.y); t[2] = f2bf(f0.z); t[3] = f2bf(f0.w);
        t[4] = f2bf(f1.x); t[5] = f2bf(f1.y); t[6] = f2bf(f1.z); t[7] = f2bf(f1.w);
        *(v8bf*)&Asl[row][half + g * 8] = t;
      }
    }
    {   // stage B tile: Bsl[c][r] = W[h][d][e], d = kt*32+r, (h,e) from col
      int r = tid >> 3;
      int cg = (tid & 7) * 16;
      int d = kt * 32 + r;
#pragma unroll
      for (int j = 0; j < 16; j += 4) {
        int C = n0 + cg + j;
        int h = C >> 6, e = C & 63;
        float4 f = *(const float4*)(W + ((size_t)h * D_ + d) * DK_ + e);
        Bsl[cg + j + 0][r] = f2bf(f.x);
        Bsl[cg + j + 1][r] = f2bf(f.y);
        Bsl[cg + j + 2][r] = f2bf(f.z);
        Bsl[cg + j + 3][r] = f2bf(f.w);
      }
    }
    __syncthreads();

    v16bf a0 = load_frag(&Asl[wm * 32 +  0][0], 40);
    v16bf a1 = load_frag(&Asl[wm * 32 + 16][0], 40);
#pragma unroll
    for (int j = 0; j < 4; ++j) {
      v16bf b = load_frag(&Bsl[wn * 64 + j * 16][0], 40);
      acc[0][j] = wmma_bf16(a0, b, acc[0][j]);
      acc[1][j] = wmma_bf16(a1, b, acc[1][j]);
    }
  }

  // epilogue: C-layout (lane = col, VGPR = row-in-halfwave) -> [B,H,S,DK] bf16
#pragma unroll
  for (int mi = 0; mi < 2; ++mi) {
#pragma unroll
    for (int j = 0; j < 4; ++j) {
#pragma unroll
      for (int r = 0; r < 8; ++r) {
        int M = m0 + wm * 32 + mi * 16 + r + ((lane >> 4) << 3);
        int C = n0 + wn * 64 + j * 16 + (lane & 15);
        int bb = M >> 11, s = M & 2047;
        int h = C >> 6, e = C & 63;
        Ot[(((size_t)bb * H_ + h) * Q_ + s) * DK_ + e] = f2bf(acc[mi][j][r] * scale);
      }
    }
  }
}

// ---------------------------------------------------------------------------
// Kernel 2: flash attention, one wave = 16 query rows, 32-key tiles.
// q/k/v: [B,H,S,64] bf16 (q pre-scaled).  mask: [B,K] bool (1 byte).
// o: [B, Q, H*DK] bf16 (head-concatenated, ready for output projection).
// ---------------------------------------------------------------------------
__global__ __launch_bounds__(256) void flash_attn_kernel(
    const __bf16* __restrict__ q_ws, const __bf16* __restrict__ k_ws,
    const __bf16* __restrict__ v_ws, const unsigned char* __restrict__ mask,
    __bf16* __restrict__ o_ws) {
  __shared__ __align__(16) __bf16 q_s[128][72];     // 128 q rows x 64 dims
  __shared__ __align__(16) __bf16 k_s[32][72];      // 32 keys  x 64 dims
  __shared__ __align__(16) __bf16 v_s[64][40];      // transposed: [dim][key]
  __shared__ __align__(16) __bf16 p_s[8][16][40];   // per-wave P staging
  __shared__ float mask_s[32];

  const int tid = threadIdx.x, lane = tid & 31, wid = tid >> 5;
  const int bh = blockIdx.y;
  const int b = bh >> 4, h = bh & 15;
  const int q0 = blockIdx.x * 128;
  const size_t head_base = ((size_t)b * H_ + h) * (size_t)K_ * DK_;

  // stage the q tile once (async DMA to LDS when available)
#pragma unroll
  for (int it = 0; it < 4; ++it) {
    int vec = tid + it * 256;
    int row = vec >> 3;
    int e0 = (vec & 7) * 8;
    const __bf16* src = q_ws + head_base + (size_t)(q0 + row) * DK_ + e0;
#if HAVE_ASYNC_LDS
    async_cp16(&q_s[row][e0], src);
#else
    *(v8bf*)&q_s[row][e0] = *(const v8bf*)src;
#endif
  }
#if HAVE_ASYNC_LDS
  wait_async0();
#endif
  __syncthreads();

  v16bf aq0 = load_frag(&q_s[wid * 16][0], 72);    // K = d 0..31
  v16bf aq1 = load_frag(&q_s[wid * 16][32], 72);   // K = d 32..63

  float m_st[8], l_st[8];
  v8f oacc[4];
#pragma unroll
  for (int r = 0; r < 8; ++r) { m_st[r] = -__builtin_inff(); l_st[r] = 0.0f; }
#pragma unroll
  for (int j = 0; j < 4; ++j) oacc[j] = (v8f)0.0f;

  for (int kt = 0; kt < K_ / 32; ++kt) {
    __syncthreads();
    {   // stage k (async, row-major) and v (sync + transpose); mask as bias
      int kr = tid >> 3;
      int e0 = (tid & 7) * 8;
      const __bf16* kp = k_ws + head_base + (size_t)(kt * 32 + kr) * DK_ + e0;
      const __bf16* vp = v_ws + head_base + (size_t)(kt * 32 + kr) * DK_ + e0;
#if HAVE_ASYNC_LDS
      async_cp16(&k_s[kr][e0], kp);
#else
      *(v8bf*)&k_s[kr][e0] = *(const v8bf*)kp;
#endif
      v8bf tv = *(const v8bf*)vp;
#pragma unroll
      for (int j = 0; j < 8; ++j) v_s[e0 + j][kr] = tv[j];
      if (tid < 32) {
        unsigned char mk = mask[b * K_ + kt * 32 + tid];
        mask_s[tid] = mk ? 0.0f : -__builtin_inff();
      }
      if (kt + 1 < K_ / 32) {      // global_prefetch_b8 for next tile
        __builtin_prefetch(kp + 32 * DK_, 0, 1);
        __builtin_prefetch(vp + 32 * DK_, 0, 1);
      }
#if HAVE_ASYNC_LDS
      wait_async0();
#endif
    }
    __syncthreads();

    // scores: 16x32 tile = two 16x16 WMMA tiles, 2 k-steps each (DK=64)
    v8f s0 = (v8f)0.0f, s1 = (v8f)0.0f;
    s0 = wmma_bf16(aq0, load_frag(&k_s[0][0], 72), s0);
    s0 = wmma_bf16(aq1, load_frag(&k_s[0][32], 72), s0);
    s1 = wmma_bf16(aq0, load_frag(&k_s[16][0], 72), s1);
    s1 = wmma_bf16(aq1, load_frag(&k_s[16][32], 72), s1);

    float bias0 = mask_s[lane & 15];
    float bias1 = mask_s[16 + (lane & 15)];

#pragma unroll
    for (int r = 0; r < 8; ++r) {
      float x0 = s0[r] + bias0;
      float x1 = s1[r] + bias1;
      float tm = fmaxf(x0, x1);
#pragma unroll
      for (int off = 8; off >= 1; off >>= 1)
        tm = fmaxf(tm, __shfl_xor(tm, off, 32));
      float mnew = fmaxf(m_st[r], tm);
      float mu = fmaxf(mnew, -1e30f);          // keep exp() args finite
      float alpha = __expf(m_st[r] - mu);      // -inf -> 0, no NaN
      float p0 = __expf(x0 - mu);
      float p1 = __expf(x1 - mu);
      float rs = p0 + p1;
#pragma unroll
      for (int off = 8; off >= 1; off >>= 1)
        rs += __shfl_xor(rs, off, 32);
      l_st[r] = l_st[r] * alpha + rs;
      m_st[r] = mnew;
      int M = r + ((lane >> 4) << 3);
      p_s[wid][M][(lane & 15)]      = f2bf(p0);
      p_s[wid][M][16 + (lane & 15)] = f2bf(p1);
#pragma unroll
      for (int j = 0; j < 4; ++j) oacc[j][r] *= alpha;
    }

    // same-wave LDS RAW: P stores -> fragment loads
    asm volatile("s_wait_dscnt 0x0" ::: "memory");

    v16bf ap = load_frag(&p_s[wid][0][0], 40);   // A = P (16 q x 32 keys)
#pragma unroll
    for (int j = 0; j < 4; ++j) {
      v16bf bv = load_frag(&v_s[j * 16][0], 40); // B = V^T (keys contract)
      oacc[j] = wmma_bf16(ap, bv, oacc[j]);
    }
  }

  // epilogue: normalize and store head-concatenated [B, Q, H*64]
#pragma unroll
  for (int r = 0; r < 8; ++r) {
    int M = r + ((lane >> 4) << 3);
    int row = q0 + wid * 16 + M;
    float inv = (l_st[r] > 0.0f) ? (1.0f / l_st[r]) : 0.0f;
#pragma unroll
    for (int j = 0; j < 4; ++j) {
      int col = h * DK_ + j * 16 + (lane & 15);
      o_ws[((size_t)b * Q_ + row) * (size_t)(H_ * DK_) + col] =
          f2bf(oacc[j][r] * inv);
    }
  }
}

// ---------------------------------------------------------------------------
// Kernel 3: output projection.  out = O[8192,1024](bf16) @ W_out[1024,1024] + b
// ---------------------------------------------------------------------------
__global__ __launch_bounds__(256) void out_proj_kernel(
    const __bf16* __restrict__ o_ws, const float* __restrict__ W_out,
    const float* __restrict__ b_out, float* __restrict__ out) {
  __shared__ __align__(16) __bf16 Asl[128][40];
  __shared__ __align__(16) __bf16 Bsl[128][40];

  const int m0 = blockIdx.x * 128;
  const int n0 = blockIdx.y * 128;
  const int tid = threadIdx.x, lane = tid & 31, wid = tid >> 5;
  const int wm = wid & 3, wn = wid >> 2;

  v8f acc[2][4];
  for (int i = 0; i < 2; ++i)
    for (int j = 0; j < 4; ++j) acc[i][j] = (v8f)0.0f;

  for (int kt = 0; kt < D_ / 32; ++kt) {
    __syncthreads();
    {   // A already bf16: async DMA straight into LDS
      int row  = tid >> 1;
      int half = (tid & 1) * 16;
      const __bf16* src = o_ws + (size_t)(m0 + row) * D_ + kt * 32 + half;
#if HAVE_ASYNC_LDS
      async_cp16(&Asl[row][half],     src);
      async_cp16(&Asl[row][half + 8], src + 8);
#else
      *(v8bf*)&Asl[row][half]     = *(const v8bf*)(src);
      *(v8bf*)&Asl[row][half + 8] = *(const v8bf*)(src + 8);
#endif
    }
    {   // B: W_out row-major [d][c] -> Bsl[c][r] (fp32 -> bf16)
      int r = tid >> 3;
      int cg = (tid & 7) * 16;
      const float* src = W_out + (size_t)(kt * 32 + r) * D_ + n0 + cg;
#pragma unroll
      for (int j = 0; j < 16; j += 4) {
        float4 f = *(const float4*)(src + j);
        Bsl[cg + j + 0][r] = f2bf(f.x);
        Bsl[cg + j + 1][r] = f2bf(f.y);
        Bsl[cg + j + 2][r] = f2bf(f.z);
        Bsl[cg + j + 3][r] = f2bf(f.w);
      }
    }
#if HAVE_ASYNC_LDS
    wait_async0();
#endif
    __syncthreads();

    v16bf a0 = load_frag(&Asl[wm * 32 +  0][0], 40);
    v16bf a1 = load_frag(&Asl[wm * 32 + 16][0], 40);
#pragma unroll
    for (int j = 0; j < 4; ++j) {
      v16bf b = load_frag(&Bsl[wn * 64 + j * 16][0], 40);
      acc[0][j] = wmma_bf16(a0, b, acc[0][j]);
      acc[1][j] = wmma_bf16(a1, b, acc[1][j]);
    }
  }

#pragma unroll
  for (int mi = 0; mi < 2; ++mi) {
#pragma unroll
    for (int j = 0; j < 4; ++j) {
#pragma unroll
      for (int r = 0; r < 8; ++r) {
        int M = m0 + wm * 32 + mi * 16 + r + ((lane >> 4) << 3);
        int C = n0 + wn * 64 + j * 16 + (lane & 15);
        out[(size_t)M * D_ + C] = acc[mi][j][r] + b_out[C];
      }
    }
  }
}

// ---------------------------------------------------------------------------
extern "C" void kernel_launch(void* const* d_in, const int* in_sizes, int n_in,
                              void* d_out, int out_size, void* d_ws,
                              size_t ws_size, hipStream_t stream) {
  const float* query = (const float*)d_in[0];
  const float* key   = (const float*)d_in[1];
  const float* value = (const float*)d_in[2];
  const unsigned char* mask = (const unsigned char*)d_in[3];
  const float* Wq    = (const float*)d_in[4];
  const float* Wk    = (const float*)d_in[5];
  const float* Wv    = (const float*)d_in[6];
  const float* W_out = (const float*)d_in[7];
  const float* b_out = (const float*)d_in[8];
  float* out = (float*)d_out;

  const size_t per = (size_t)B_ * H_ * Q_ * DK_;   // 8,388,608 bf16 elements
  __bf16* q_ws = (__bf16*)d_ws;
  __bf16* k_ws = q_ws + per;
  __bf16* v_ws = k_ws + per;
  __bf16* o_ws = v_ws + per;                       // total 64 MB of workspace

  qkv_proj_kernel<<<dim3(64, 8, 3), 256, 0, stream>>>(
      query, key, value, Wq, Wk, Wv, q_ws, k_ws, v_ws);
  flash_attn_kernel<<<dim3(16, 64), 256, 0, stream>>>(
      q_ws, k_ws, v_ws, mask, o_ws);
  out_proj_kernel<<<dim3(64, 8), 256, 0, stream>>>(o_ws, W_out, b_out, out);
}
